// HybridCNNRNN_9620726743468
// MI455X (gfx1250) — compile-verified, matrix operand order checked by reading, not run on hardware
//
#include <hip/hip_runtime.h>
#include <hip/hip_bf16.h>
#include <stdint.h>

// ---------------------------------------------------------------------------
// Model dims (compile-time)
// ---------------------------------------------------------------------------
constexpr int Tt  = 100;   // time steps
constexpr int Bb  = 512;   // batch
constexpr int ICn = 21;    // input channels
constexpr int NSn = 7;     // strikes (width)
constexpr int CFn = 64;    // conv filters
constexpr int Hh  = 128;   // lstm hidden
constexpr int Gn  = 512;   // 4*H gates
constexpr int K0  = CFn * NSn;  // 448  layer-0 lstm input dim
constexpr int K1  = 2 * Hh;     // 256  layer-1 lstm input dim
constexpr int Mseq = Tt * Bb;   // 51200 rows in the time-batched sequence

constexpr int MB   = 16;        // batch rows per recurrent block
constexpr int NBLK = Bb / MB;   // 32 recurrent blocks per direction

typedef __attribute__((ext_vector_type(16))) _Float16 v16h;
typedef __attribute__((ext_vector_type(8)))  float    v8f;
typedef __attribute__((ext_vector_type(4)))  unsigned int u32x4;
typedef __attribute__((ext_vector_type(8)))  int      i32x8;
typedef __attribute__((ext_vector_type(4)))  int      i32x4;

#if defined(__has_builtin)
#if __has_builtin(__builtin_amdgcn_tensor_load_to_lds)
#define HAVE_TDM 1
#endif
#endif
#ifndef HAVE_TDM
#define HAVE_TDM 0
#endif

// ---------------------------------------------------------------------------
// WMMA helpers (wave32, v_wmma_f32_16x16x32_f16)
// ---------------------------------------------------------------------------
__device__ inline v8f wmma_f16(v16h a, v16h b, v8f c) {
  return __builtin_amdgcn_wmma_f32_16x16x32_f16(
      /*neg_a=*/false, a, /*neg_b=*/false, b,
      /*c_mod=*/(short)0, c, /*reuse_a=*/false, /*reuse_b=*/false);
}

// Load a 16x32 (rows x K) f16 fragment from a row-major (rows, ldk) matrix.
// Works for both A (rows = M tile) and B (weights stored (N,K) row-major,
// so a B fragment for column block n0 is the same gather with rows = n0..n0+15).
// Per the CDNA5 ISA layout: lane l<16 holds row r=l, K = kbase+[0..7] and
// kbase+[16..23]; lane l>=16 holds row r=l-16, K = kbase+[8..15], +[24..31].
// Each lane: two contiguous 16-byte loads.
__device__ inline v16h load_frag16(const _Float16* __restrict__ base,
                                   int row0, int ldk, int kbase) {
  const int lane = threadIdx.x & 31;
  const int r    = lane & 15;
  const int hi   = (lane >> 4) & 1;
  const _Float16* p = base + (size_t)(row0 + r) * (size_t)ldk + kbase + hi * 8;
  union { v16h v; uint32_t u[8]; } f;
  const uint32_t* a = reinterpret_cast<const uint32_t*>(p);
  const uint32_t* b = reinterpret_cast<const uint32_t*>(p + 16);
#pragma unroll
  for (int i = 0; i < 4; ++i) { f.u[i] = a[i]; f.u[4 + i] = b[i]; }
  return f.v;
}

// C/D 16x16 f32 tile layout: VGPR r, lane<16 -> (M=r,   N=lane),
//                            VGPR r, lane>=16 -> (M=r+8, N=lane-16)
__device__ inline void store_tile_g(float* __restrict__ out, int ld,
                                    int m0, int n0, v8f acc, float bias) {
  const int lane = threadIdx.x & 31;
  const int col  = n0 + (lane & 15);
  const int r0   = m0 + ((lane >> 4) ? 8 : 0);
#pragma unroll
  for (int r = 0; r < 8; ++r)
    out[(size_t)(r0 + r) * (size_t)ld + col] = acc[r] + bias;
}

__device__ inline void store_tile_lds(float* __restrict__ out, int ld,
                                      int m0, int n0, v8f acc) {
  const int lane = threadIdx.x & 31;
  const int col  = n0 + (lane & 15);
  const int r0   = m0 + ((lane >> 4) ? 8 : 0);
#pragma unroll
  for (int r = 0; r < 8; ++r)
    out[(r0 + r) * ld + col] = acc[r];
}

__device__ inline float sigf(float x) { return 1.0f / (1.0f + __expf(-x)); }

// ---------------------------------------------------------------------------
// TDM: async 1-D copy of nElems f32 from global -> LDS (single-wave issue).
// Descriptor packing per CDNA5 ISA D# groups 0/1 (1-D tile, data_size=4B).
// ---------------------------------------------------------------------------
#if HAVE_TDM
__device__ inline void tdm_load_1d_f32(void* lds_dst, const void* gsrc,
                                       unsigned nElems) {
  u32x4 g0;
  i32x8 g1;
  i32x4 g2 = (i32x4)0, g3 = (i32x4)0;
  const unsigned lds_addr = (unsigned)(size_t)lds_dst;       // LDS byte addr
  const unsigned long long ga = (unsigned long long)(size_t)gsrc;
  g0[0] = 1u;                                   // count=1, user descriptor
  g0[1] = lds_addr;                             // lds_addr [63:32]
  g0[2] = (unsigned)ga;                         // global_addr [95:64]
  g0[3] = (unsigned)((ga >> 32) & 0x01FFFFFFull) | (2u << 30); // addr hi | type=2
  g1[0] = (int)(2u << 16);                      // data_size=4B, wg_mask=0
  g1[1] = (int)((nElems & 0xFFFFu) << 16);      // tensor_dim0[15:0]
  g1[2] = (int)((nElems >> 16) | (1u << 16));   // tensor_dim0 hi | tensor_dim1=1
  g1[3] = (int)((nElems & 0xFFFFu) << 16);      // tile_dim0 = nElems
  g1[4] = 0;                                    // tile_dim1/2 = 0 (1-D tile)
  g1[5] = (int)nElems;                          // tensor_dim0_stride lo
  g1[6] = 0;
  g1[7] = 0;
#if __clang_major__ >= 23
  __builtin_amdgcn_tensor_load_to_lds(g0, g1, g2, g3, (i32x8)0, 0);
#else
  __builtin_amdgcn_tensor_load_to_lds(g0, g1, g2, g3, 0);
#endif
}
#endif

// ---------------------------------------------------------------------------
// Kernel: f32 -> f16 weight conversion
// ---------------------------------------------------------------------------
__global__ void k_convert(const float* __restrict__ src,
                          _Float16* __restrict__ dst, int n) {
  int i = blockIdx.x * blockDim.x + threadIdx.x;
  if (i < n) dst[i] = (_Float16)src[i];
}

// Kernel: bias sum (bih + bhh)
__global__ void k_biassum(const float* __restrict__ a,
                          const float* __restrict__ b,
                          float* __restrict__ out, int n) {
  int i = blockIdx.x * blockDim.x + threadIdx.x;
  if (i < n) out[i] = a[i] + b[i];
}

// ---------------------------------------------------------------------------
// Kernel: fused conv1+BN+ReLU -> conv2+BN+ReLU for one (b,t) slab.
// Output written in the (T, B, CF*NS) layout used by the LSTM GEMMs, as f16.
// ---------------------------------------------------------------------------
__global__ void __launch_bounds__(64)
k_cnn(const float* __restrict__ x,
      const float* __restrict__ c1w, const float* __restrict__ c1b,
      const float* __restrict__ g1, const float* __restrict__ b1,
      const float* __restrict__ m1, const float* __restrict__ v1,
      const float* __restrict__ c2w, const float* __restrict__ c2b,
      const float* __restrict__ g2, const float* __restrict__ b2,
      const float* __restrict__ m2, const float* __restrict__ v2,
      _Float16* __restrict__ seq16) {
  __shared__ float in_l[ICn][NSn + 2];   // zero-padded width
  __shared__ float mid_l[CFn][NSn + 2];
  const int bt = blockIdx.x;             // = b*T + t  (x is (B,T,IC,NS))
  const int b = bt / Tt, t = bt % Tt;
  const float* xp = x + (size_t)bt * (ICn * NSn);
  const int tid = threadIdx.x;

  for (int i = tid; i < ICn * (NSn + 2); i += 64) {
    int ic = i / (NSn + 2), w = i % (NSn + 2) - 1;
    in_l[ic][w + 1] = (w >= 0 && w < NSn) ? xp[ic * NSn + w] : 0.0f;
  }
  __syncthreads();

  const int cf = tid;  // 64 threads == CF
  // conv1 + BN + ReLU
  {
    const float inv  = g1[cf] * rsqrtf(v1[cf] + 1e-5f);
    const float beta = b1[cf] - m1[cf] * inv;
    const float* wrow = c1w + cf * (ICn * 3);
    mid_l[cf][0] = 0.0f; mid_l[cf][NSn + 1] = 0.0f;
#pragma unroll
    for (int ns = 0; ns < NSn; ++ns) {
      float s = c1b[cf];
      for (int ic = 0; ic < ICn; ++ic) {
#pragma unroll
        for (int k = 0; k < 3; ++k) s += wrow[ic * 3 + k] * in_l[ic][ns + k];
      }
      float y = s * inv + beta;
      mid_l[cf][ns + 1] = fmaxf(y, 0.0f);
    }
  }
  __syncthreads();
  // conv2 + BN + ReLU -> seq16 (t, b, cf*7+ns)
  {
    const float inv  = g2[cf] * rsqrtf(v2[cf] + 1e-5f);
    const float beta = b2[cf] - m2[cf] * inv;
    const float* wrow = c2w + cf * (CFn * 3);
    _Float16* op = seq16 + ((size_t)t * Bb + b) * K0 + cf * NSn;
#pragma unroll
    for (int ns = 0; ns < NSn; ++ns) {
      float s = c2b[cf];
      for (int ic = 0; ic < CFn; ++ic) {
#pragma unroll
        for (int k = 0; k < 3; ++k) s += wrow[ic * 3 + k] * mid_l[ic][ns + k];
      }
      float y = s * inv + beta;
      op[ns] = (_Float16)fmaxf(y, 0.0f);
    }
  }
}

// ---------------------------------------------------------------------------
// Kernel: WMMA GEMM  out(M,512) = A(M,K) @ W(512,K)^T + biassum
// Block = 256 threads (8 waves), 128x64 output tile, 2x2 tiles per wave
// (4 WMMA per 2 A-frag + 2 B-frag loads -> 2x operand reuse vs 1x2 tiling).
// ---------------------------------------------------------------------------
__global__ void __launch_bounds__(256)
k_gemm(const _Float16* __restrict__ A, const _Float16* __restrict__ W,
       const float* __restrict__ bias, float* __restrict__ out,
       int M, int K) {
  const int tid  = threadIdx.x;
  const int wave = tid >> 5;
  const int m0   = blockIdx.x * 128;
  const int n0   = blockIdx.y * 64;
  const int mt   = (wave & 3) * 2;       // 2 M-tiles per wave (of 8)
  const int nt   = (wave >> 2) * 2;      // 2 N-tiles per wave (of 4)
  v8f acc00 = {}, acc01 = {}, acc10 = {}, acc11 = {};
  const int nkb = K >> 5;
  for (int kb = 0; kb < nkb; ++kb) {
    v16h a0 = load_frag16(A, m0 + mt * 16, K, kb * 32);
    v16h a1 = load_frag16(A, m0 + (mt + 1) * 16, K, kb * 32);
    v16h b0 = load_frag16(W, n0 + nt * 16, K, kb * 32);
    v16h b1 = load_frag16(W, n0 + (nt + 1) * 16, K, kb * 32);
    acc00 = wmma_f16(a0, b0, acc00);
    acc01 = wmma_f16(a0, b1, acc01);
    acc10 = wmma_f16(a1, b0, acc10);
    acc11 = wmma_f16(a1, b1, acc11);
  }
  const int lane = tid & 31;
  const float bv0 = bias[n0 + nt * 16 + (lane & 15)];
  const float bv1 = bias[n0 + (nt + 1) * 16 + (lane & 15)];
  store_tile_g(out, Gn, m0 + mt * 16,       n0 + nt * 16,       acc00, bv0);
  store_tile_g(out, Gn, m0 + mt * 16,       n0 + (nt + 1) * 16, acc01, bv1);
  store_tile_g(out, Gn, m0 + (mt + 1) * 16, n0 + nt * 16,       acc10, bv0);
  store_tile_g(out, Gn, m0 + (mt + 1) * 16, n0 + (nt + 1) * 16, acc11, bv1);
}

// ---------------------------------------------------------------------------
// Kernel: persistent bidirectional-LSTM recurrence, BOTH directions fused
// into one launch (blockIdx.y = direction) -> 64 concurrent workgroups.
// One block owns 16 batch rows for all T steps. whh (512x128 f16) lives in
// LDS; per step: gates(16x512) = h(16x128) @ whh^T via WMMA, while the TDM
// streams the NEXT step's x-projection chunk (16x512 f32 = 32 KB, contiguous)
// into the other half of a double buffer (s_wait_tensorcnt(1) pipelining);
// then fused gate nonlinearity + state update, h re-encoded to f16 in LDS.
// Dynamic LDS: 128K (whh) + 4K (h) + 2x32K (xp dbl buf) + 32K (gates) = 228 KB.
// ---------------------------------------------------------------------------
constexpr size_t LSTM_LDS_WHH   = (size_t)Gn * Hh * 2;      // 131072
constexpr size_t LSTM_LDS_H     = (size_t)MB * Hh * 2;      //   4096
constexpr size_t LSTM_LDS_XP    = (size_t)MB * Gn * 4;      //  32768 (x2)
constexpr size_t LSTM_LDS_GATES = (size_t)MB * Gn * 4;      //  32768
constexpr size_t LSTM_LDS_BYTES =
    LSTM_LDS_WHH + LSTM_LDS_H + 2 * LSTM_LDS_XP + LSTM_LDS_GATES;

__global__ void __launch_bounds__(256)
k_lstm(const float* __restrict__ xprojF,    // (T, B, 512) f32, bias folded in
       const float* __restrict__ xprojB,
       const _Float16* __restrict__ whhF16, // (512, 128) f16
       const _Float16* __restrict__ whhB16,
       _Float16* __restrict__ hout,         // (T, B, outStride) f16
       int outStride) {
  extern __shared__ float4 smem4[];
  char* smem = reinterpret_cast<char*>(smem4);
  _Float16* whh_l = reinterpret_cast<_Float16*>(smem);
  _Float16* h_l   = reinterpret_cast<_Float16*>(smem + LSTM_LDS_WHH);
  float*    xp0   = reinterpret_cast<float*>(smem + LSTM_LDS_WHH + LSTM_LDS_H);
  float*    xp1   = reinterpret_cast<float*>(smem + LSTM_LDS_WHH + LSTM_LDS_H + LSTM_LDS_XP);
  float*    g_l   = reinterpret_cast<float*>(smem + LSTM_LDS_WHH + LSTM_LDS_H + 2 * LSTM_LDS_XP);

  const int tid  = threadIdx.x;
  const int wave = tid >> 5;
  const int dir  = blockIdx.y;             // 0 = forward, 1 = backward
  const int b0   = blockIdx.x * MB;
  const int colOff = dir * Hh;
  const float* __restrict__ xproj = dir ? xprojB : xprojF;
  const _Float16* __restrict__ whh16 = dir ? whhB16 : whhF16;

  // Stage whh into LDS (16B chunks)
  {
    const uint4* src = reinterpret_cast<const uint4*>(whh16);
    uint4* dst = reinterpret_cast<uint4*>(whh_l);
    for (int i = tid; i < (int)(LSTM_LDS_WHH / 16); i += blockDim.x) dst[i] = src[i];
  }
  for (int i = tid; i < MB * Hh; i += blockDim.x) h_l[i] = (_Float16)0.0f;

  float c[8];
#pragma unroll
  for (int k = 0; k < 8; ++k) c[k] = 0.0f;

#if HAVE_TDM
  if (tid == 0) {
    const int t0 = dir ? (Tt - 1) : 0;
    tdm_load_1d_f32(xp0, xproj + ((size_t)t0 * Bb + b0) * Gn,
                    (unsigned)(MB * Gn));
  }
#endif
  __syncthreads();

  // 8 waves, single 16-row M tile; each wave owns 4 N-tiles: wave + 8*j
  for (int step = 0; step < Tt; ++step) {
    const int t = dir ? (Tt - 1 - step) : step;
    float* xp_cur = (step & 1) ? xp1 : xp0;
    float* xp_nxt = (step & 1) ? xp0 : xp1;

#if HAVE_TDM
    if (tid == 0 && step + 1 < Tt) {      // prefetch next step via TDM
      const int tn = dir ? (Tt - 2 - step) : (step + 1);
      tdm_load_1d_f32(xp_nxt, xproj + ((size_t)tn * Bb + b0) * Gn,
                      (unsigned)(MB * Gn));
    }
#else
    (void)xp_nxt;
    {
      const float* xsrc = xproj + ((size_t)t * Bb + b0) * Gn;
      for (int i = tid; i < MB * Gn; i += blockDim.x) xp_cur[i] = xsrc[i];
    }
#endif

    // gates = h @ whh^T   (overlaps with the in-flight TDM transfers)
    v16h aF[4];
#pragma unroll
    for (int kb = 0; kb < 4; ++kb)
      aF[kb] = load_frag16(h_l, 0, Hh, kb * 32);
#pragma unroll
    for (int j = 0; j < 4; ++j) {
      const int nt = wave + 8 * j;
      v8f acc = {};
#pragma unroll
      for (int kb = 0; kb < 4; ++kb) {
        v16h bF = load_frag16(whh_l, nt * 16, Hh, kb * 32);
        acc = wmma_f16(aF[kb], bF, acc);
      }
      store_tile_lds(g_l, Gn, 0, nt * 16, acc);
    }

#if HAVE_TDM
    if (tid == 0) {
      if (step + 1 < Tt) __builtin_amdgcn_s_wait_tensorcnt(1);  // cur done
      else               __builtin_amdgcn_s_wait_tensorcnt(0);
    }
#endif
    __syncthreads();

    // Fused gate nonlinearity + cell/hidden update (PyTorch i,f,g,o order)
#pragma unroll
    for (int k = 0; k < 8; ++k) {
      const int cell = tid + 256 * k;   // 16*128 = 2048 cells
      const int m  = cell >> 7;
      const int j2 = cell & 127;
      const float gi = g_l[m * Gn + j2]        + xp_cur[m * Gn + j2];
      const float gf = g_l[m * Gn + 128 + j2]  + xp_cur[m * Gn + 128 + j2];
      const float gg = g_l[m * Gn + 256 + j2]  + xp_cur[m * Gn + 256 + j2];
      const float go = g_l[m * Gn + 384 + j2]  + xp_cur[m * Gn + 384 + j2];
      const float cv = sigf(gf) * c[k] + sigf(gi) * tanhf(gg);
      c[k] = cv;
      const float hv = sigf(go) * tanhf(cv);
      h_l[m * Hh + j2] = (_Float16)hv;
      hout[((size_t)t * Bb + b0 + m) * outStride + colOff + j2] = (_Float16)hv;
    }
    __syncthreads();
  }
}

// ---------------------------------------------------------------------------
// Kernel: head  out(B,2) = relu(last(B,256) @ fc1^T + b1) @ fc2^T + b2
// ---------------------------------------------------------------------------
__global__ void __launch_bounds__(64)
k_head(const _Float16* __restrict__ last,   // (B, 256) f16
       const float* __restrict__ fc1w, const float* __restrict__ fc1b,
       const float* __restrict__ fc2w, const float* __restrict__ fc2b,
       float* __restrict__ out) {
  __shared__ float h1[64];
  const int b = blockIdx.x;
  const int j = threadIdx.x;
  const _Float16* lp = last + (size_t)b * K1;
  float s = fc1b[j];
  for (int k = 0; k < K1; ++k) s += (float)lp[k] * fc1w[j * K1 + k];
  h1[j] = fmaxf(s, 0.0f);
  __syncthreads();
  if (j < 2) {
    float s2 = fc2b[j];
#pragma unroll
    for (int k = 0; k < 64; ++k) s2 += h1[k] * fc2w[j * 64 + k];
    out[(size_t)b * 2 + j] = s2;
  }
}

// ---------------------------------------------------------------------------
// Host launcher
// ---------------------------------------------------------------------------
extern "C" void kernel_launch(void* const* d_in, const int* in_sizes, int n_in,
                              void* d_out, int out_size, void* d_ws, size_t ws_size,
                              hipStream_t stream) {
  (void)in_sizes; (void)n_in; (void)out_size; (void)ws_size;

  const float* x   = (const float*)d_in[0];
  const float* c1w = (const float*)d_in[1];  const float* c1b = (const float*)d_in[2];
  const float* g1  = (const float*)d_in[3];  const float* b1  = (const float*)d_in[4];
  const float* m1  = (const float*)d_in[5];  const float* v1  = (const float*)d_in[6];
  const float* c2w = (const float*)d_in[7];  const float* c2b = (const float*)d_in[8];
  const float* g2  = (const float*)d_in[9];  const float* b2  = (const float*)d_in[10];
  const float* m2  = (const float*)d_in[11]; const float* v2  = (const float*)d_in[12];
  const float* wih0f = (const float*)d_in[13]; const float* whh0f = (const float*)d_in[14];
  const float* bih0f = (const float*)d_in[15]; const float* bhh0f = (const float*)d_in[16];
  const float* wih0b = (const float*)d_in[17]; const float* whh0b = (const float*)d_in[18];
  const float* bih0b = (const float*)d_in[19]; const float* bhh0b = (const float*)d_in[20];
  const float* wih1f = (const float*)d_in[21]; const float* whh1f = (const float*)d_in[22];
  const float* bih1f = (const float*)d_in[23]; const float* bhh1f = (const float*)d_in[24];
  const float* wih1b = (const float*)d_in[25]; const float* whh1b = (const float*)d_in[26];
  const float* bih1b = (const float*)d_in[27]; const float* bhh1b = (const float*)d_in[28];
  const float* fc1w = (const float*)d_in[29]; const float* fc1b = (const float*)d_in[30];
  const float* fc2w = (const float*)d_in[31]; const float* fc2b = (const float*)d_in[32];
  float* out = (float*)d_out;

  // Workspace carve-out
  size_t off = 0;
  auto take = [&](size_t bytes) -> void* {
    off = (off + 255) & ~(size_t)255;
    void* p = (char*)d_ws + off;
    off += bytes;
    return p;
  };
  _Float16* seq16 = (_Float16*)take((size_t)Mseq * K0 * 2);   // (T,B,448) f16
  float* xpf      = (float*)take((size_t)Mseq * Gn * 4);      // (T,B,512) f32
  float* xpb      = (float*)take((size_t)Mseq * Gn * 4);
  _Float16* l0out = (_Float16*)take((size_t)Mseq * K1 * 2);   // (T,B,256) f16
  _Float16* l1out = (_Float16*)take((size_t)Mseq * K1 * 2);
  _Float16* wih0f16 = (_Float16*)take((size_t)Gn * K0 * 2);
  _Float16* wih0b16 = (_Float16*)take((size_t)Gn * K0 * 2);
  _Float16* whh0f16 = (_Float16*)take((size_t)Gn * Hh * 2);
  _Float16* whh0b16 = (_Float16*)take((size_t)Gn * Hh * 2);
  _Float16* wih1f16 = (_Float16*)take((size_t)Gn * K1 * 2);
  _Float16* wih1b16 = (_Float16*)take((size_t)Gn * K1 * 2);
  _Float16* whh1f16 = (_Float16*)take((size_t)Gn * Hh * 2);
  _Float16* whh1b16 = (_Float16*)take((size_t)Gn * Hh * 2);
  float* bs0f = (float*)take(Gn * 4);
  float* bs0b = (float*)take(Gn * 4);
  float* bs1f = (float*)take(Gn * 4);
  float* bs1b = (float*)take(Gn * 4);

  auto cvt = [&](const float* s, _Float16* d, int n) {
    k_convert<<<(n + 255) / 256, 256, 0, stream>>>(s, d, n);
  };
  cvt(wih0f, wih0f16, Gn * K0); cvt(wih0b, wih0b16, Gn * K0);
  cvt(whh0f, whh0f16, Gn * Hh); cvt(whh0b, whh0b16, Gn * Hh);
  cvt(wih1f, wih1f16, Gn * K1); cvt(wih1b, wih1b16, Gn * K1);
  cvt(whh1f, whh1f16, Gn * Hh); cvt(whh1b, whh1b16, Gn * Hh);
  k_biassum<<<2, 256, 0, stream>>>(bih0f, bhh0f, bs0f, Gn);
  k_biassum<<<2, 256, 0, stream>>>(bih0b, bhh0b, bs0b, Gn);
  k_biassum<<<2, 256, 0, stream>>>(bih1f, bhh1f, bs1f, Gn);
  k_biassum<<<2, 256, 0, stream>>>(bih1b, bhh1b, bs1b, Gn);

  // CNN -> (T,B,448) f16 sequence
  k_cnn<<<Bb * Tt, 64, 0, stream>>>(x, c1w, c1b, g1, b1, m1, v1,
                                    c2w, c2b, g2, b2, m2, v2, seq16);

  dim3 ggrid(Mseq / 128, Gn / 64);

  // Layer 0: input projections (big WMMA GEMMs), then fused dual-direction
  // persistent recurrence (64 workgroups live during the serial phase).
  k_gemm<<<ggrid, 256, 0, stream>>>(seq16, wih0f16, bs0f, xpf, Mseq, K0);
  k_gemm<<<ggrid, 256, 0, stream>>>(seq16, wih0b16, bs0b, xpb, Mseq, K0);

  (void)hipFuncSetAttribute(reinterpret_cast<const void*>(k_lstm),
                            hipFuncAttributeMaxDynamicSharedMemorySize,
                            (int)LSTM_LDS_BYTES);
  k_lstm<<<dim3(NBLK, 2), 256, LSTM_LDS_BYTES, stream>>>(
      xpf, xpb, whh0f16, whh0b16, l0out, K1);

  // Layer 1
  k_gemm<<<ggrid, 256, 0, stream>>>(l0out, wih1f16, bs1f, xpf, Mseq, K1);
  k_gemm<<<ggrid, 256, 0, stream>>>(l0out, wih1b16, bs1b, xpb, Mseq, K1);
  k_lstm<<<dim3(NBLK, 2), 256, LSTM_LDS_BYTES, stream>>>(
      xpf, xpb, whh1f16, whh1b16, l1out, K1);

  // Head on seq[-1] = l1out[T-1]
  const _Float16* last = l1out + (size_t)(Tt - 1) * Bb * K1;
  k_head<<<Bb, 64, 0, stream>>>(last, fc1w, fc1b, fc2w, fc2b, out);
}
